// SelfAttentionHead_18442589569535
// MI455X (gfx1250) — compile-verified, matrix operand order checked by reading, not run on hardware
//
#include <hip/hip_runtime.h>

typedef __attribute__((ext_vector_type(16))) __bf16 v16bf;
typedef __attribute__((ext_vector_type(8)))  float  v8f;
typedef __attribute__((ext_vector_type(8)))  int    v8i;

#define Bn 8
#define Tn 2048
#define Dn 1024
#define Hn 64

__device__ __forceinline__ unsigned short f2bf(float f) {
  unsigned int u = __float_as_uint(f);
  u += 0x7FFFu + ((u >> 16) & 1u);          // round-to-nearest-even
  return (unsigned short)(u >> 16);
}

// K index of the p-th 2-element pair in a 16-bit A/B WMMA fragment (16x32).
__device__ __forceinline__ int pairK(int lane, int p) {
  return ((lane >> 4) << 3) + ((p < 4) ? (p << 1) : ((p << 1) + 8));
}

__device__ __forceinline__ v8f wmma_bf16(v8i a, v8i b, v8f c) {
  return __builtin_amdgcn_wmma_f32_16x16x32_bf16(
      false, __builtin_bit_cast(v16bf, a),
      false, __builtin_bit_cast(v16bf, b),
      (short)0, c, false, false);
}

// ---------------------------------------------------------------------------
// Kernel 1: K/Q/V projections. 64 rows x 192 cols per workgroup (8 waves).
// K projection is pre-scaled by head_size^-0.5 = 0.125.
// ---------------------------------------------------------------------------
__global__ __launch_bounds__(256) void proj_kernel(
    const float* __restrict__ x, const float* __restrict__ Wk,
    const float* __restrict__ Wq, const float* __restrict__ Wv,
    unsigned short* __restrict__ Kb, unsigned short* __restrict__ Qb,
    unsigned short* __restrict__ Vb) {
  __shared__ unsigned short ldsX[64 * 72];    // x chunk, row-major [row][k]
  __shared__ unsigned short ldsW[192 * 72];   // weights transposed [col][k]

  const int tid  = threadIdx.x;
  const int lane = tid & 31;
  const int w    = tid >> 5;
  const int n    = lane & 15;
  const int h    = (lane >> 4) & 1;
  const int rw   = w & 3;    // row tile 0..3 (16 rows each)
  const int cg   = w >> 2;   // col group 0..1 (6 col tiles each)
  const int row0 = blockIdx.x * 64;

  const float* Ws[3] = {Wk, Wq, Wv};

  v8f acc[6] = {};

  for (int kb = 0; kb < Dn; kb += 64) {
    // stage x[row0..+64, kb..+64] as bf16
    {
      int r  = tid >> 2;
      int k0 = (tid & 3) << 4;
      const float* src = x + (size_t)(row0 + r) * Dn + kb + k0;
      #pragma unroll
      for (int k = 0; k < 16; ++k) ldsX[r * 72 + k0 + k] = f2bf(src[k]);
      if (kb + 64 < Dn) __builtin_prefetch(src + 64, 0, 1);  // next chunk
    }
    // stage Wk|Wq|Wv chunk transposed: ldsW[wid*64+n][k]
    for (int i = 0; i < 48; ++i) {
      int e = i * 256 + tid;
      int wid = e >> 12;
      int rem = e & 4095;
      int k  = rem >> 6;
      int nn = rem & 63;
      ldsW[(wid * 64 + nn) * 72 + k] = f2bf(Ws[wid][(size_t)(kb + k) * Hn + nn]);
    }
    __syncthreads();

    const unsigned int* lX = (const unsigned int*)ldsX;
    const unsigned int* lW = (const unsigned int*)ldsW;
    for (int kk = 0; kk < 64; kk += 32) {
      v8i a;
      int arow = 16 * rw + n;
      #pragma unroll
      for (int p = 0; p < 8; ++p)
        a[p] = (int)lX[(arow * 72 + kk + pairK(lane, p)) >> 1];
      #pragma unroll
      for (int t = 0; t < 6; ++t) {
        int col = (cg * 6 + t) * 16 + n;
        v8i bfr;
        #pragma unroll
        for (int p = 0; p < 8; ++p)
          bfr[p] = (int)lW[(col * 72 + kk + pairK(lane, p)) >> 1];
        acc[t] = wmma_bf16(a, bfr, acc[t]);
      }
    }
    __syncthreads();
  }

  #pragma unroll
  for (int t = 0; t < 6; ++t) {
    int ct   = cg * 6 + t;
    int wid  = ct >> 2;
    int ncol = (ct & 3) * 16 + n;
    unsigned short* dst = (wid == 0) ? Kb : (wid == 1) ? Qb : Vb;
    float sc = (wid == 0) ? 0.125f : 1.0f;
    #pragma unroll
    for (int r = 0; r < 8; ++r) {
      int row = row0 + 16 * rw + r + 8 * h;
      dst[(size_t)row * Hn + ncol] = f2bf(acc[t][r] * sc);
    }
  }
}

// ---------------------------------------------------------------------------
// Kernel 2: causal flash attention. wei = (0.125*K) @ Q^T, softmax, @ V.
// One wave owns 16 output rows; 128 rows per workgroup. The j-loop is
// workgroup-uniform so Q/V blocks are cooperatively staged in LDS
// (Q via async-to-LDS, V transposed for B-fragment reads).
// ---------------------------------------------------------------------------
__global__ __launch_bounds__(256) void attn_kernel(
    const unsigned short* __restrict__ Kb, const unsigned short* __restrict__ Qb,
    const unsigned short* __restrict__ Vb, float* __restrict__ out) {
  __shared__ unsigned short ldsQ[32 * 64];       // Q block, row-major [j][h]
  __shared__ unsigned short ldsVt[64 * 40];      // V block transposed [h][j]
  __shared__ unsigned short ldsP[8 * 16 * 40];   // per-wave P transpose buffer

  const int tid  = threadIdx.x;
  const int lane = tid & 31;
  const int w    = tid >> 5;
  const int n    = lane & 15;
  const int h    = (lane >> 4) & 1;

  const int b       = blockIdx.x >> 4;          // 16 row-blocks per batch
  const int i0      = (blockIdx.x & 15) * 128;
  const int rowbase = i0 + 16 * w;

  const unsigned int*   K32 = (const unsigned int*)Kb + (size_t)b * Tn * (Hn / 2);
  const unsigned short* Qp  = Qb + (size_t)b * Tn * Hn;
  const unsigned short* Vp  = Vb + (size_t)b * Tn * Hn;

  // resident A fragments of this wave's (pre-scaled) K rows: 16 x 64
  v8i ka0, ka1;
  {
    int arow = rowbase + n;
    #pragma unroll
    for (int p = 0; p < 8; ++p) {
      int kk = pairK(lane, p);
      ka0[p] = (int)K32[(arow * Hn + kk) >> 1];
      ka1[p] = (int)K32[(arow * Hn + 32 + kk) >> 1];
    }
  }

  float m[8], l[8];
  v8f o[4] = {};
  #pragma unroll
  for (int r = 0; r < 8; ++r) { m[r] = -1e30f; l[r] = 0.0f; }

  unsigned short*     myP   = ldsP + w * (16 * 40);
  const unsigned int* myP32 = (const unsigned int*)myP;
  const unsigned int* lQ    = (const unsigned int*)ldsQ;
  const unsigned int* lVt   = (const unsigned int*)ldsVt;

  const int jmax = i0 + 127;                    // uniform across the block
  for (int jb = 0; jb <= jmax; jb += 32) {
    // ---- cooperative staging ----
    {
      // Q rows [jb, jb+32) x 64 -> ldsQ, async DMA (16 B per thread)
      int row = tid >> 3, seg = tid & 7;
      const unsigned short* src = Qp + (size_t)(jb + row) * Hn + seg * 8;
      unsigned ldso = (unsigned)(uintptr_t)(&ldsQ[row * 64 + seg * 8]);
      asm volatile("global_load_async_to_lds_b128 %0, %1, off"
                   :: "v"(ldso), "v"(src) : "memory");
    }
    {
      // V rows [jb, jb+32) x 64, transposed -> ldsVt[h][j]
      int j = tid >> 3, h0 = (tid & 7) * 8;
      const uint4 vv = *(const uint4*)(Vp + (size_t)(jb + j) * Hn + h0);
      const unsigned int vals[4] = {vv.x, vv.y, vv.z, vv.w};
      #pragma unroll
      for (int e = 0; e < 4; ++e) {
        ldsVt[(h0 + 2 * e    ) * 40 + j] = (unsigned short)(vals[e] & 0xFFFFu);
        ldsVt[(h0 + 2 * e + 1) * 40 + j] = (unsigned short)(vals[e] >> 16);
      }
    }
    asm volatile("s_wait_asynccnt 0" ::: "memory");
    __syncthreads();

    if (jb <= rowbase + 15) {                   // wave-uniform predicate
      // ---- S = K Q^T tiles (16x16 each), K-dim = 64 ----
      v8f s0 = {}, s1 = {};
      {
        int qrow = n;                           // tile s=0
        v8i q0, q1;
        #pragma unroll
        for (int p = 0; p < 8; ++p) {
          int kk = pairK(lane, p);
          q0[p] = (int)lQ[(qrow * 64 + kk) >> 1];
          q1[p] = (int)lQ[(qrow * 64 + 32 + kk) >> 1];
        }
        s0 = wmma_bf16(ka0, q0, s0);
        s0 = wmma_bf16(ka1, q1, s0);
      }
      const bool act1 = (jb + 16) <= (rowbase + 15);
      if (act1) {
        int qrow = 16 + n;                      // tile s=1
        v8i q0, q1;
        #pragma unroll
        for (int p = 0; p < 8; ++p) {
          int kk = pairK(lane, p);
          q0[p] = (int)lQ[(qrow * 64 + kk) >> 1];
          q1[p] = (int)lQ[(qrow * 64 + 32 + kk) >> 1];
        }
        s1 = wmma_bf16(ka0, q0, s1);
        s1 = wmma_bf16(ka1, q1, s1);
      }

      // ---- causal mask + online softmax (rows live in 16-lane halves) ----
      float p0[8], p1[8], scl[8];
      #pragma unroll
      for (int r = 0; r < 8; ++r) {
        int M = rowbase + 8 * h + r;
        float v0 = ((jb + n) <= M) ? s0[r] : -1e30f;
        float v1 = (act1 && (jb + 16 + n) <= M) ? s1[r] : -1e30f;
        float rm = fmaxf(v0, v1);
        for (int msk = 1; msk < 16; msk <<= 1)
          rm = fmaxf(rm, __shfl_xor(rm, msk, 32));
        float mn = fmaxf(m[r], rm);
        float sc = __expf(m[r] - mn);
        float e0 = __expf(v0 - mn);
        float e1 = __expf(v1 - mn);
        float rs = e0 + e1;
        for (int msk = 1; msk < 16; msk <<= 1)
          rs += __shfl_xor(rs, msk, 32);
        m[r] = mn;
        l[r] = l[r] * sc + rs;
        scl[r] = sc;
        p0[r] = e0;
        p1[r] = e1;
      }
      #pragma unroll
      for (int t = 0; t < 4; ++t)
        #pragma unroll
        for (int r = 0; r < 8; ++r) o[t][r] *= scl[r];

      // ---- C-layout P -> A-layout via per-wave LDS bounce ----
      #pragma unroll
      for (int r = 0; r < 8; ++r) {
        int prow = r + 8 * h;
        myP[prow * 40 + n]      = f2bf(p0[r]);
        myP[prow * 40 + 16 + n] = f2bf(p1[r]);
      }
      asm volatile("s_wait_dscnt 0" ::: "memory");
      v8i pa;
      #pragma unroll
      for (int p = 0; p < 8; ++p)
        pa[p] = (int)myP32[(n * 40 + pairK(lane, p)) >> 1];

      // ---- O += P @ V  (V B-fragments from transposed LDS) ----
      #pragma unroll
      for (int t = 0; t < 4; ++t) {
        int hcol = 16 * t + n;
        v8i vf;
        #pragma unroll
        for (int p = 0; p < 8; ++p)
          vf[p] = (int)lVt[(hcol * 40 + pairK(lane, p)) >> 1];
        o[t] = wmma_bf16(pa, vf, o[t]);
      }
    }
    __syncthreads();
  }

  // ---- normalize + store fp32 ----
  #pragma unroll
  for (int t = 0; t < 4; ++t)
    #pragma unroll
    for (int r = 0; r < 8; ++r) {
      int row = rowbase + r + 8 * h;
      int col = 16 * t + n;
      out[((size_t)(b * Tn + row)) * Hn + col] = o[t][r] / l[r];
    }
}

extern "C" void kernel_launch(void* const* d_in, const int* in_sizes, int n_in,
                              void* d_out, int out_size, void* d_ws, size_t ws_size,
                              hipStream_t stream) {
  const float* x  = (const float*)d_in[0];
  const float* Wk = (const float*)d_in[1];
  const float* Wq = (const float*)d_in[2];
  const float* Wv = (const float*)d_in[3];

  // workspace: bf16 K, Q, V  (3 * 16384 * 64 * 2 bytes = 6 MB)
  unsigned short* Kb = (unsigned short*)d_ws;
  unsigned short* Qb = Kb + (size_t)Bn * Tn * Hn;
  unsigned short* Vb = Qb + (size_t)Bn * Tn * Hn;

  proj_kernel<<<(Bn * Tn) / 64, 256, 0, stream>>>(x, Wk, Wq, Wv, Kb, Qb, Vb);
  attn_kernel<<<Bn * (Tn / 128), 256, 0, stream>>>(Kb, Qb, Vb, (float*)d_out);
}
